// PatternMixer_5059471475268
// MI455X (gfx1250) — matrix operand order repaired and checked by reading, not compiled, not run on hardware
//
#include <hip/hip_runtime.h>
#include <hip/hip_bf16.h>

// ---------------------------------------------------------------------------
// PatternMixer for MI455X (gfx1250, wave32).
//
// Dominant work: C = relu(A @ W^T + b), M=735 (pad 768), K=1024, N=8192.
// 12.3 GFLOP vs ~200 MB total traffic -> bf16 WMMA (v_wmma_f32_16x16x32_bf16,
// 16384 FLOP/instr, f32 accumulate) keeps the GEMM near the 23.3 TB/s
// bandwidth roofline; fp32 WMMA (16x16x4) would be ~8x more matrix issues.
// Final 8192x57x57 output write (106 MB) is the hard bandwidth floor (~4.6us).
// ---------------------------------------------------------------------------

#define K_DIM     1024        // NUM_BASIC
#define N_DIM     8192        // NUM_MIXED
#define M_ROWS    735         // T(15) * 7 * 7
#define M_PAD     768
#define N_OUT     57
#define OUT_PER_J (N_OUT * N_OUT)   // 3249

typedef __attribute__((ext_vector_type(16))) __bf16 v16bf;
typedef __attribute__((ext_vector_type(8)))  float  v8f;

union FragU {            // two 16B loads -> one 16-element bf16 WMMA fragment
  uint4 q[2];
  v16bf v;
};

__device__ __forceinline__ unsigned short f2bf(float f) {
  unsigned u = __builtin_bit_cast(unsigned, f);
  u += 0x7FFFu + ((u >> 16) & 1u);           // round-to-nearest-even
  return (unsigned short)(u >> 16);
}

// ---------------------------------------------------------------------------
// Kernel 1: A[r,h] = relu((1+sigmoid(te[h,t])) * mat[h,n,m]), r = t*49+m*7+n.
// Rows 735..767 zero-padded so the GEMM needs no M bounds checks.
// ---------------------------------------------------------------------------
__global__ __launch_bounds__(256)
void build_A(const float* __restrict__ mat, const float* __restrict__ tex,
             unsigned short* __restrict__ Abf) {
  int e = blockIdx.x * 256 + threadIdx.x;    // 768*1024 elements
  int r = e >> 10;
  int h = e & 1023;
  float v = 0.0f;
  if (r < M_ROWS) {
    int t   = r / 49;
    int rem = r - t * 49;
    int m   = rem / 7;
    int n   = rem - m * 7;
    float te = 1.0f + 1.0f / (1.0f + __expf(-tex[h * 15 + t]));
    float x  = te * mat[h * 49 + n * 7 + m];
    v = x > 0.0f ? x : 0.0f;
  }
  Abf[e] = f2bf(v);
}

// ---------------------------------------------------------------------------
// Kernel 2: W (8192x1024 f32) -> bf16, vectorized.
// ---------------------------------------------------------------------------
__global__ __launch_bounds__(256)
void convert_W(const float4* __restrict__ W4, ushort4* __restrict__ Wb4) {
  size_t e = (size_t)blockIdx.x * 256 + threadIdx.x;   // 2,097,152 float4s
  float4 f = W4[e];
  ushort4 o;
  o.x = f2bf(f.x); o.y = f2bf(f.y); o.z = f2bf(f.z); o.w = f2bf(f.w);
  Wb4[e] = o;
}

// ---------------------------------------------------------------------------
// Kernel 3: C[r,j] = relu(sum_h A[r,h]*W[j,h] + b[j]), r-major store.
// Workgroup tile 128(M) x 128(N); 8 waves (2x4); wave tile 64x32 = 4x2
// accumulators of v_wmma_f32_16x16x32_bf16.
//
// Fragment layouts per CDNA5 ISA 7.12.2:
//   A 16x32 bf16: lane L<16 -> row M=L, K {0..7} in v0-3, {16..23} in v4-7;
//                 lane L>=16 -> row M=L-16, K {8..15} / {24..31}.
//   B 32x16 bf16: lane L<16 -> col N=L, K {0..7} / {16..23};
//                 lane L>=16 -> col N=L-16, K {8..15} / {24..31}.
// W is j-major/K-contiguous, so B columns are W rows -> straight loads.
// ---------------------------------------------------------------------------
__global__ __launch_bounds__(256)
void gemm_wmma(const unsigned short* __restrict__ A,
               const unsigned short* __restrict__ W,
               const float* __restrict__ bias,
               float* __restrict__ C) {
  const int tid  = threadIdx.x;
  const int lane = tid & 31;
  const int wave = tid >> 5;           // 0..7
  const int wy   = wave >> 2;          // 0..1  (M)
  const int wx   = wave & 3;           // 0..3  (N)

  const int waveM = blockIdx.y * 128 + wy * 64;   // grid.y = 6  -> 768 rows
  const int waveN = blockIdx.x * 128 + wx * 32;   // grid.x = 64 -> 8192 cols

  const int lrow   = lane & 15;
  const int hi16   = lane >> 4;                   // 0 or 1
  const int kA     = hi16 << 3;                   // A frag: +0 / +8
  const int kB     = hi16 << 3;                   // B frag: +0 / +8

  v8f acc[4][2] = {};

  for (int kk = 0; kk < K_DIM; kk += 32) {
    // ---- B fragments (2 N-tiles), two 16B chunks per lane (K kB..+7, kB+16..+23) ----
    FragU bf[2];
#pragma unroll
    for (int nt = 0; nt < 2; ++nt) {
      const unsigned short* src =
          W + (size_t)(waveN + nt * 16 + lrow) * K_DIM + kk + kB;
      const uint4* p = (const uint4*)src;
      bf[nt].q[0] = p[0];
      bf[nt].q[1] = p[2];
      if (kk + 32 < K_DIM)                         // global_prefetch_b8
        __builtin_prefetch((const void*)(src + 32), 0, 0);
    }
    // ---- A fragments (4 M-tiles) + WMMAs ----
#pragma unroll
    for (int mt = 0; mt < 4; ++mt) {
      const unsigned short* src =
          A + (size_t)(waveM + mt * 16 + lrow) * K_DIM + kk + kA;
      FragU af;
      af.q[0] = *(const uint4*)src;                // K kA .. kA+7
      af.q[1] = *(const uint4*)(src + 16);         // K kA+16 .. kA+23
      acc[mt][0] = __builtin_amdgcn_wmma_f32_16x16x32_bf16(
          false, af.v, false, bf[0].v, (short)0, acc[mt][0], false, false);
      acc[mt][1] = __builtin_amdgcn_wmma_f32_16x16x32_bf16(
          false, af.v, false, bf[1].v, (short)0, acc[mt][1], false, false);
    }
  }

  // Epilogue: +bias, relu, r-major store (lanes 0..15 contiguous in j).
  // C/D layout: VGPR i, lane L -> M = i + (L>=16 ? 8 : 0), N = L&15.
#pragma unroll
  for (int nt = 0; nt < 2; ++nt) {
    const int   col = waveN + nt * 16 + lrow;
    const float bv  = bias[col];
#pragma unroll
    for (int mt = 0; mt < 4; ++mt) {
      const int rbase = waveM + mt * 16 + (hi16 << 3);
      float* dst = C + (size_t)rbase * N_DIM + col;
#pragma unroll
      for (int i = 0; i < 8; ++i) {
        float v = acc[mt][nt][i] + bv;
        dst[(size_t)i * N_DIM] = v > 0.0f ? v : 0.0f;
      }
    }
  }
}

// ---------------------------------------------------------------------------
// Kernel 4: per-j (8192 blocks) build the 57x57 normalized block.
//   mixed[1+p, 1+q] = C[(7*(7-c)+q)*7+n, j],  p = c*7+n
//   border ones at (0, 0::7) and (0::7, 0); d = rsqrt(clip(rowsum,1)).
// C column staged in LDS (C is L2-resident: 24 MB << 192 MB L2).
// ---------------------------------------------------------------------------
__global__ __launch_bounds__(256)
void finalize(const float* __restrict__ C, float* __restrict__ out) {
  __shared__ float rowC[736];
  __shared__ float dsh[N_OUT];
  const int j = blockIdx.x;

  for (int r = threadIdx.x; r < M_ROWS; r += 256)
    rowC[r] = C[(size_t)r * N_DIM + j];
  __syncthreads();

  if (threadIdx.x < N_OUT) {
    float s;
    if (threadIdx.x == 0) {
      s = 9.0f;                                   // ones at cols 0,7,...,56
    } else {
      int p = threadIdx.x - 1;
      int c = p / 7;
      int n = p - c * 7;
      int base = (7 * (7 - c)) * 7 + n;           // q=0; +7 per q
      s = (n == 6) ? 1.0f : 0.0f;                 // column-0 border one
      for (int q = 0; q < 56; ++q) s += rowC[base + q * 7];
    }
    s = s < 1.0f ? 1.0f : s;
    dsh[threadIdx.x] = rsqrtf(s);
  }
  __syncthreads();

  float* o = out + (size_t)j * OUT_PER_J;
  for (int idx = threadIdx.x; idx < OUT_PER_J; idx += 256) {
    int row = idx / N_OUT;
    int col = idx - row * N_OUT;
    float v;
    if (row == 0) {
      v = (col % 7 == 0) ? 1.0f : 0.0f;           // includes (0,0)
    } else if (col == 0) {
      v = (row % 7 == 0) ? 1.0f : 0.0f;
    } else {
      int p = row - 1;
      int c = p / 7;
      int n = p - c * 7;
      v = rowC[(7 * (7 - c) + (col - 1)) * 7 + n];
    }
    o[idx] = dsh[row] * v * dsh[col];
  }
}

// ---------------------------------------------------------------------------
extern "C" void kernel_launch(void* const* d_in, const int* in_sizes, int n_in,
                              void* d_out, int out_size, void* d_ws, size_t ws_size,
                              hipStream_t stream) {
  const float* mat = (const float*)d_in[0];   // 1024*7*7
  const float* tex = (const float*)d_in[1];   // 1024*15
  const float* W   = (const float*)d_in[2];   // 8192*1024
  const float* b   = (const float*)d_in[3];   // 8192
  float* out       = (float*)d_out;           // 8192*57*57

  // Workspace layout (total ~41.5 MB):
  //   [0, 1.5MB)        A  bf16  768x1024
  //   [1.5MB, 17.5MB)   W  bf16  8192x1024
  //   [17.5MB, 41.5MB)  C  f32   768x8192 (r-major)
  char* ws = (char*)d_ws;
  unsigned short* Abf = (unsigned short*)ws;
  unsigned short* Wbf = (unsigned short*)(ws + (size_t)1572864);
  float*          Cws = (float*)(ws + (size_t)1572864 + (size_t)16777216);

  build_A<<<(M_PAD * K_DIM) / 256, 256, 0, stream>>>(mat, tex, Abf);
  convert_W<<<(N_DIM * K_DIM / 4) / 256, 256, 0, stream>>>(
      (const float4*)W, (ushort4*)Wbf);
  gemm_wmma<<<dim3(N_DIM / 128, M_PAD / 128), 256, 0, stream>>>(Abf, Wbf, b, Cws);
  finalize<<<N_DIM, 256, 0, stream>>>(Cws, out);
}
